// _DSSL_10376640987773
// MI455X (gfx1250) — matrix operand order repaired and checked by reading, not compiled, not run on hardware
//
#include <hip/hip_runtime.h>

// ---------------- types ----------------
typedef __attribute__((ext_vector_type(16))) _Float16 v16h;
typedef __attribute__((ext_vector_type(8)))  _Float16 v8h;
typedef __attribute__((ext_vector_type(8)))  float    v8f;
typedef __attribute__((ext_vector_type(4)))  unsigned int v4u;
typedef __attribute__((ext_vector_type(8)))  int      v8i_t;
typedef __attribute__((ext_vector_type(4)))  int      v4i_t;

// ---------------- problem constants ----------------
#define B_    4096
#define N_    10
#define D_    64
#define H_    256
#define F_    128
#define E0_   90      // N*(N-1) edges per batch, row-major (i, j!=i)
#define ROWS_E 96     // 90 padded to 6 tiles of 16
#define MT_E  6

// ---------------- f16 weight workspace layout (in halves), all [N][K] K-contiguous ----------------
#define WS_EW0 0          // edge_w0^T [256][128]
#define WS_EW1 32768      // edge_w1^T [256][256]
#define WS_EW2 98304      // edge_w2^T [256][256]
#define WS_NW0 163840     // node_w0^T [256][448]
#define WS_NW1 278528     // node_w1^T [256][256]
#define WS_NW2 344064     // node_w2^T [64][256]

// ---------------- LDS layout (bytes), total 231424 <= 320 KB/WGP ----------------
#define OFF_W     0            // staged edge weights, max 256*256*2 = 131072
#define OFF_EA    131072       // edge act ping  [96][256] f16 (E_in uses [96][128])
#define OFF_EB    180224       // edge act pong  [96][256] f16
#define OFF_NODES 229376       // node states f16 [10][64] (+pad)
#define SMEM_BYTES 231424
// node-stage overlays inside the (dead) EA region:
#define OFF_AGG  (OFF_EA)          // f32 [16][256]  = 16384 B
#define OFF_NIN  (OFF_EA + 16384)  // f16 [16][448]  = 14336 B
#define OFF_NA   (OFF_EA + 30720)  // f16 [16][256]  =  8192 B
#define OFF_NB   (OFF_EA + 38912)  // f16 [16][256]  =  8192 B

#if __has_builtin(__builtin_amdgcn_tensor_load_to_lds) && __has_builtin(__builtin_amdgcn_s_wait_tensorcnt)
#define HAS_TDM 1
#else
#define HAS_TDM 0
#endif

// ---------------- TDM: DMA a 2D f16 tile [Nrows][K] (K-contiguous) from global to LDS ----------------
__device__ __forceinline__ void tdm_issue_w(const _Float16* gsrc, unsigned ldsOff,
                                            unsigned K, unsigned Nrows) {
#if HAS_TDM
  unsigned long long ga = (unsigned long long)(size_t)gsrc;
  v4u g0;
  g0.x = 1u;                                            // count=1, user descriptor
  g0.y = ldsOff;                                        // lds_addr (bytes)
  g0.z = (unsigned)(ga & 0xffffffffull);                // global_addr[31:0]
  g0.w = (unsigned)((ga >> 32) & 0x1ffffffull) | (2u << 30); // addr[56:32] | type=2
  v8i_t g1;
  g1[0] = (int)(1u << 16);                              // data_size = 2 bytes
  g1[1] = (int)((K & 0xffffu) << 16);                   // tensor_dim0[15:0] -> bits 63:48
  g1[2] = (int)(((K >> 16) & 0xffffu) | ((Nrows & 0xffffu) << 16)); // dim0 hi | dim1 lo
  g1[3] = (int)(((Nrows >> 16) & 0xffffu) | ((K & 0xffffu) << 16)); // dim1 hi | tile_dim0=K
  g1[4] = (int)(Nrows & 0xffffu);                       // tile_dim1 = Nrows (tile_dim2=0)
  g1[5] = (int)K;                                       // tensor_dim0_stride low32
  g1[6] = 0; g1[7] = 0;
  v4i_t z4 = {0, 0, 0, 0};
  v8i_t z8 = {0, 0, 0, 0, 0, 0, 0, 0};
  // 6-arg form (clang-23 / therock-10.0 headers): (g0, g1, g2, g3, extra, cpol)
  __builtin_amdgcn_tensor_load_to_lds(g0, g1, z4, z4, z8, 0);
#else
  (void)gsrc; (void)ldsOff; (void)K; (void)Nrows;
#endif
}

__device__ __forceinline__ void copy_w_fallback(_Float16* dst, const _Float16* src,
                                                int nHalves, int tid) {
  const uint4* s = (const uint4*)src;
  uint4* d = (uint4*)dst;
  for (int i = tid; i < nHalves / 8; i += 256) d[i] = s[i];
}

// ---------------- WMMA fragment containers ----------------
union AFrag { v16h v; v8h h[2]; };

// GEMM over 16x16 tiles: D = A[M x 32KT] * W^T + bias, epilogue lambda per tile.
// A rows in `inL` (stride inStride halves); W^T rows [N][K] in `w` (LDS or global).
template <int KT, class Epi>
__device__ __forceinline__ void gemm_f16(const _Float16* __restrict__ inL, int inStride,
                                         const _Float16* __restrict__ w, int wStride,
                                         const float* __restrict__ bias,
                                         int Mtiles, int Ntiles, int wave, int lane, Epi epi) {
  const int laneN = lane & 15;
  const int hi    = lane >> 4;   // 0: lanes 0-15, 1: lanes 16-31
  for (int mt = 0; mt < Mtiles; ++mt) {
    // A fragment layout (16-bit 16x32): lane M = lane%16;
    // lanes<16 hold K [0..7]+[16..23], lanes>=16 hold K [8..15]+[24..31]
    AFrag a[KT];
    const _Float16* arow = inL + (mt * 16 + laneN) * inStride + hi * 8;
#pragma unroll
    for (int kt = 0; kt < KT; ++kt) {
      a[kt].h[0] = *(const v8h*)(arow + kt * 32);
      a[kt].h[1] = *(const v8h*)(arow + kt * 32 + 16);
    }
    for (int nt = wave; nt < Ntiles; nt += 8) {
      const float bv = bias[nt * 16 + laneN];
      v8f acc = {bv, bv, bv, bv, bv, bv, bv, bv};
      // B fragment layout (32x16): lane N = lane%16; lanes<16 K [0..7]+[16..23], lanes>=16 K [8..15]+[24..31]
      const _Float16* wrow = w + (nt * 16 + laneN) * wStride + hi * 8;
#pragma unroll
      for (int kt = 0; kt < KT; ++kt) {
        AFrag bf;
        bf.h[0] = *(const v8h*)(wrow + kt * 32);
        bf.h[1] = *(const v8h*)(wrow + kt * 32 + 16);
        acc = __builtin_amdgcn_wmma_f32_16x16x32_f16(false, a[kt].v, false, bf.v,
                                                     (short)0, acc, false, false);
      }
      epi(mt, nt, acc);
    }
  }
}

// Row LayerNorm (+ReLU) over H_=256 columns, f16 in-place, wave32 shuffle reduction.
__device__ __forceinline__ void ln_relu(_Float16* buf, int rows,
                                        const float* __restrict__ g,
                                        const float* __restrict__ be,
                                        int wave, int lane) {
  const int h0 = lane * 8;
  float gq[8], bq[8];
#pragma unroll
  for (int q = 0; q < 8; ++q) { gq[q] = g[h0 + q]; bq[q] = be[h0 + q]; }
  for (int r = wave; r < rows; r += 8) {
    float x[8], s = 0.f, s2 = 0.f;
#pragma unroll
    for (int q = 0; q < 8; ++q) {
      x[q] = (float)buf[r * H_ + h0 + q];
      s += x[q]; s2 += x[q] * x[q];
    }
#pragma unroll
    for (int off = 16; off > 0; off >>= 1) {
      s  += __shfl_xor(s,  off, 32);
      s2 += __shfl_xor(s2, off, 32);
    }
    const float mean = s * (1.f / 256.f);
    const float var  = fmaxf(s2 * (1.f / 256.f) - mean * mean, 0.f);
    const float rstd = rsqrtf(var + 1e-5f);
#pragma unroll
    for (int q = 0; q < 8; ++q) {
      float y = (x[q] - mean) * rstd * gq[q] + bq[q];
      buf[r * H_ + h0 + q] = (_Float16)fmaxf(y, 0.f);
    }
  }
}

// ---------------- f32 -> f16 transposed weight conversion: dst[n*K+k] = src[k*N+n] ----------------
__global__ void wconv_kernel(const float* __restrict__ src, _Float16* __restrict__ dst,
                             int K, int N) {
  int idx = blockIdx.x * blockDim.x + threadIdx.x;
  if (idx < K * N) {
    int n = idx / K, k = idx % K;
    dst[idx] = (_Float16)src[k * N + n];
  }
}

// ---------------- fused per-batch graph-net kernel ----------------
__global__ __launch_bounds__(256, 1)
void fused_gn_kernel(const float* __restrict__ states, const int* __restrict__ feature,
                     const float* __restrict__ eb0, const float* __restrict__ eb1,
                     const float* __restrict__ eg,  const float* __restrict__ ebe,
                     const float* __restrict__ eb2,
                     const float* __restrict__ nb0, const float* __restrict__ nb1,
                     const float* __restrict__ ng,  const float* __restrict__ nbe,
                     const float* __restrict__ nb2,
                     const _Float16* __restrict__ wts, float* __restrict__ out) {
  extern __shared__ char smem[];
  _Float16* sW   = (_Float16*)(smem + OFF_W);
  _Float16* sEA  = (_Float16*)(smem + OFF_EA);
  _Float16* sEB  = (_Float16*)(smem + OFF_EB);
  _Float16* sNd  = (_Float16*)(smem + OFF_NODES);
  float*    sAgg = (float*)   (smem + OFF_AGG);
  _Float16* sNin = (_Float16*)(smem + OFF_NIN);
  _Float16* sNA  = (_Float16*)(smem + OFF_NA);
  _Float16* sNB  = (_Float16*)(smem + OFF_NB);

  const int b = blockIdx.x;
  const int tid = threadIdx.x;
  const int wave = tid >> 5, lane = tid & 31;
  const int laneN = lane & 15, hi = lane >> 4;

  // ---- stage edge_w0 via TDM while we build inputs ----
#if HAS_TDM
  if (wave == 0) tdm_issue_w(wts + WS_EW0, OFF_W, 128, 256);
#endif

  // node states -> f16 LDS
  for (int i = tid; i < N_ * D_; i += 256) sNd[i] = (_Float16)states[b * (N_ * D_) + i];
  __syncthreads();

  // E_in [96][128]: row e = concat(node[i], node[j]); topology is fixed (i, j!=i row-major)
  for (int idx = tid; idx < ROWS_E * 128; idx += 256) {
    int e = idx >> 7, c = idx & 127;
    _Float16 v = (_Float16)0.f;
    if (e < E0_) {
      int i = e / 9, q = e % 9;
      int j = q + (q >= i ? 1 : 0);
      v = (c < 64) ? sNd[i * 64 + c] : sNd[j * 64 + (c - 64)];
    }
    sEA[idx] = v;  // stride 128
  }
#if HAS_TDM
  if (wave == 0) __builtin_amdgcn_s_wait_tensorcnt(0);
#else
  copy_w_fallback(sW, wts + WS_EW0, 128 * 256, tid);
#endif
  __syncthreads();

  // ---- edge layer 0: [96x128] @ [128x256] + b0, ReLU -> sEB ----
  gemm_f16<4>(sEA, 128, sW, 128, eb0, MT_E, 16, wave, lane,
    [&](int mt, int nt, v8f acc) {
#pragma unroll
      for (int r = 0; r < 8; ++r)
        sEB[(mt * 16 + r + 8 * hi) * H_ + nt * 16 + laneN] = (_Float16)fmaxf(acc[r], 0.f);
    });

  // stage edge_w1
  __syncthreads();
#if HAS_TDM
  if (wave == 0) { tdm_issue_w(wts + WS_EW1, OFF_W, 256, 256); __builtin_amdgcn_s_wait_tensorcnt(0); }
#else
  copy_w_fallback(sW, wts + WS_EW1, 256 * 256, tid);
#endif
  __syncthreads();

  // ---- edge layer 1: [96x256] @ [256x256] + b1 -> sEA (pre-LN) ----
  gemm_f16<8>(sEB, 256, sW, 256, eb1, MT_E, 16, wave, lane,
    [&](int mt, int nt, v8f acc) {
#pragma unroll
      for (int r = 0; r < 8; ++r)
        sEA[(mt * 16 + r + 8 * hi) * H_ + nt * 16 + laneN] = (_Float16)acc[r];
    });
  __syncthreads();

  // stage edge_w2 overlapped with LayerNorm
#if HAS_TDM
  if (wave == 0) tdm_issue_w(wts + WS_EW2, OFF_W, 256, 256);
#endif
  ln_relu(sEA, ROWS_E, eg, ebe, wave, lane);
#if HAS_TDM
  if (wave == 0) __builtin_amdgcn_s_wait_tensorcnt(0);
#else
  __syncthreads();
  copy_w_fallback(sW, wts + WS_EW2, 256 * 256, tid);
#endif
  __syncthreads();

  // ---- edge layer 2: [96x256] @ [256x256] + b2 -> sEB (edge_attr) ----
  gemm_f16<8>(sEA, 256, sW, 256, eb2, MT_E, 16, wave, lane,
    [&](int mt, int nt, v8f acc) {
#pragma unroll
      for (int r = 0; r < 8; ++r)
        sEB[(mt * 16 + r + 8 * hi) * H_ + nt * 16 + laneN] = (_Float16)acc[r];
    });
  __syncthreads();

  // ---- segment-sum: node i aggregates its 9 edges (rows 9i..9i+8) ----
  for (int idx = tid; idx < N_ * H_; idx += 256) {
    int n = idx >> 8, h = idx & 255;
    float s = 0.f;
#pragma unroll
    for (int q = 0; q < 9; ++q) s += (float)sEB[(n * 9 + q) * H_ + h];
    sAgg[idx] = s;
  }
  __syncthreads();

  // ---- node input [16][448] = concat(node[64], one_hot[128], agg[256]) ----
  const int fv = feature[b];
  for (int idx = tid; idx < 16 * 448; idx += 256) {
    int m = idx / 448, c = idx % 448;
    _Float16 v = (_Float16)0.f;
    if (m < N_) {
      if (c < 64)       v = sNd[m * 64 + c];
      else if (c < 192) v = (_Float16)((c - 64 == fv) ? 1.f : 0.f);
      else              v = (_Float16)sAgg[m * H_ + (c - 192)];
    }
    sNin[idx] = v;
  }
  __syncthreads();

  // ---- node layer 0: [16x448] @ [448x256] + b0, ReLU (weights streamed from L2: M=16, no reuse) ----
  gemm_f16<14>(sNin, 448, wts + WS_NW0, 448, nb0, 1, 16, wave, lane,
    [&](int mt, int nt, v8f acc) {
#pragma unroll
      for (int r = 0; r < 8; ++r)
        sNA[(r + 8 * hi) * H_ + nt * 16 + laneN] = (_Float16)fmaxf(acc[r], 0.f);
    });
  __syncthreads();

  // ---- node layer 1: [16x256] @ [256x256] + b1 -> sNB (pre-LN) ----
  gemm_f16<8>(sNA, 256, wts + WS_NW1, 256, nb1, 1, 16, wave, lane,
    [&](int mt, int nt, v8f acc) {
#pragma unroll
      for (int r = 0; r < 8; ++r)
        sNB[(r + 8 * hi) * H_ + nt * 16 + laneN] = (_Float16)acc[r];
    });
  __syncthreads();

  ln_relu(sNB, 16, ng, nbe, wave, lane);
  __syncthreads();

  // ---- node layer 2: [16x256] @ [256x64] + b2, residual, write f32 out ----
  gemm_f16<8>(sNB, 256, wts + WS_NW2, 256, nb2, 1, 4, wave, lane,
    [&](int mt, int nt, v8f acc) {
#pragma unroll
      for (int r = 0; r < 8; ++r) {
        int m = r + 8 * hi;
        if (m < N_) {
          int gi = b * (N_ * D_) + m * D_ + nt * 16 + laneN;
          out[gi] = states[gi] + acc[r];
        }
      }
    });
}

// ---------------- host launcher ----------------
extern "C" void kernel_launch(void* const* d_in, const int* in_sizes, int n_in,
                              void* d_out, int out_size, void* d_ws, size_t ws_size,
                              hipStream_t stream) {
  (void)in_sizes; (void)n_in; (void)out_size; (void)ws_size;
  const float* states = (const float*)d_in[0];
  const int*   feat   = (const int*)d_in[1];
  // d_in[2] edge_index: unused — topology is deterministic (fully connected, no self loops)
  const float* ew0 = (const float*)d_in[3];
  const float* eb0 = (const float*)d_in[4];
  const float* ew1 = (const float*)d_in[5];
  const float* eb1 = (const float*)d_in[6];
  const float* eg  = (const float*)d_in[7];
  const float* ebe = (const float*)d_in[8];
  const float* ew2 = (const float*)d_in[9];
  const float* eb2 = (const float*)d_in[10];
  const float* nw0 = (const float*)d_in[11];
  const float* nb0 = (const float*)d_in[12];
  const float* nw1 = (const float*)d_in[13];
  const float* nb1 = (const float*)d_in[14];
  const float* ng  = (const float*)d_in[15];
  const float* nbe = (const float*)d_in[16];
  const float* nw2 = (const float*)d_in[17];
  const float* nb2 = (const float*)d_in[18];
  _Float16* wts = (_Float16*)d_ws;

  // convert weights to f16, transposed [N][K] K-contiguous (L2-resident, done once per call)
  struct { const float* s; int off, K, N; } cv[6] = {
    {ew0, WS_EW0, 128, 256}, {ew1, WS_EW1, 256, 256}, {ew2, WS_EW2, 256, 256},
    {nw0, WS_NW0, 448, 256}, {nw1, WS_NW1, 256, 256}, {nw2, WS_NW2, 256, 64},
  };
  for (int i = 0; i < 6; ++i) {
    int total = cv[i].K * cv[i].N;
    wconv_kernel<<<(total + 255) / 256, 256, 0, stream>>>(cv[i].s, wts + cv[i].off,
                                                          cv[i].K, cv[i].N);
  }

  static_assert(SMEM_BYTES <= 320 * 1024, "exceeds WGP LDS");
  (void)hipFuncSetAttribute(reinterpret_cast<const void*>(fused_gn_kernel),
                            hipFuncAttributeMaxDynamicSharedMemorySize, SMEM_BYTES);
  fused_gn_kernel<<<B_, 256, SMEM_BYTES, stream>>>(
      states, feat, eb0, eb1, eg, ebe, eb2, nb0, nb1, ng, nbe, nb2, wts, (float*)d_out);
}
